// NTXentLoss_7911329760051
// MI455X (gfx1250) — compile-verified, matrix operand order checked by reading, not run on hardware
//
#include <hip/hip_runtime.h>
#include <hip/hip_bf16.h>
#include <stdint.h>

typedef __attribute__((ext_vector_type(16))) _Float16 v16h;
typedef __attribute__((ext_vector_type(8)))  float    v8f;

#define N_TOT  8192
#define DIM    256
#define BHALF  4096
#define INV_T  2.0f
#define EPSN   1e-8f

#define ROWS_PER_BLOCK 128
#define COL_CHUNKS 8
#define TILES_PER_CHUNK (N_TOT / 16 / COL_CHUNKS)   /* 64 */
#define LDS_STRIDE (DIM + 8)   /* f16 elems; 528B row -> conflict-free, 16B aligned */

// exp(logit - 2) with logit = c * INV_T, done in base-2 (v_exp_f32 is exp2):
// exp2( c * INV_T*log2e - 2*log2e )
#define EXP2_SCALE 2.8853900817779268f   /* 2 * log2(e) */
#define EXP2_BIAS (-2.8853900817779268f) /* -2 * log2(e) */

union FragH { v16h h; uint32_t u[8]; };
union Pack8 { uint4 q; _Float16 h[8]; };

// ---------------------------------------------------------------------------
// Kernel 1: row-normalize [z1; z2] (f32) -> zn (f16, row-major). 1 wave / row.
// ---------------------------------------------------------------------------
__global__ void __launch_bounds__(256)
ntx_normalize(const float* __restrict__ z1, const float* __restrict__ z2,
              _Float16* __restrict__ zn) {
    const int row  = blockIdx.x * (blockDim.x >> 5) + (threadIdx.x >> 5);
    const int lane = threadIdx.x & 31;
    const float* src = (row < BHALF) ? (z1 + (size_t)row * DIM)
                                     : (z2 + (size_t)(row - BHALF) * DIM);
    float4 a = ((const float4*)src)[lane * 2 + 0];
    float4 b = ((const float4*)src)[lane * 2 + 1];
    float ss = a.x*a.x + a.y*a.y + a.z*a.z + a.w*a.w
             + b.x*b.x + b.y*b.y + b.z*b.z + b.w*b.w;
    for (int off = 16; off; off >>= 1) ss += __shfl_xor(ss, off, 32);
    const float inv = 1.0f / fmaxf(sqrtf(ss), EPSN);
    Pack8 p;
    p.h[0] = (_Float16)(a.x * inv); p.h[1] = (_Float16)(a.y * inv);
    p.h[2] = (_Float16)(a.z * inv); p.h[3] = (_Float16)(a.w * inv);
    p.h[4] = (_Float16)(b.x * inv); p.h[5] = (_Float16)(b.y * inv);
    p.h[6] = (_Float16)(b.z * inv); p.h[7] = (_Float16)(b.w * inv);
    ((uint4*)(zn + (size_t)row * DIM))[lane] = p.q;
}

// ---------------------------------------------------------------------------
// Kernel 2: positive logits pos[i] = dot(zn_i, zn_{(i+B)%N}) / T.  1 wave/row.
// ---------------------------------------------------------------------------
__global__ void __launch_bounds__(256)
ntx_positives(const _Float16* __restrict__ zn, float* __restrict__ pos_logit) {
    const int row  = blockIdx.x * (blockDim.x >> 5) + (threadIdx.x >> 5);
    const int lane = threadIdx.x & 31;
    const int partner = (row < BHALF) ? (row + BHALF) : (row - BHALF);
    Pack8 x, y;
    x.q = ((const uint4*)(zn + (size_t)row * DIM))[lane];
    y.q = ((const uint4*)(zn + (size_t)partner * DIM))[lane];
    float d = 0.0f;
    #pragma unroll
    for (int e = 0; e < 8; ++e) d += (float)x.h[e] * (float)y.h[e];
    for (int off = 16; off; off >>= 1) d += __shfl_xor(d, off, 32);
    if (lane == 0) pos_logit[row] = d * INV_T;
}

// ---------------------------------------------------------------------------
// Kernel 3: tiled Z*Z^T via v_wmma_f32_16x16x32_f16 with streaming sum of
// exp(logit-2). Grid (row-blocks x col-chunks). Block = 8 waves = 128 rows,
// chunk = 1024 columns. B tiles staged to LDS with global_load_async_to_lds
// (ASYNCcnt path). The hot loop is branch-free; the diagonal term is
// subtracted once at the end (self-dot recomputed from resident A frags).
// Partial sums land in psum[chunk][row] (no atomics -> deterministic).
// ---------------------------------------------------------------------------
__global__ void __launch_bounds__(256)
ntx_sim_lse(const _Float16* __restrict__ zn, float* __restrict__ psum) {
    __shared__ _Float16 ldsB[16 * LDS_STRIDE];

    const int tid  = threadIdx.x;
    const int lane = tid & 31;
    const int wave = tid >> 5;
    const int hh   = lane >> 4;   // half-wave: K sub-block / row offset
    const int lm   = lane & 15;
    const int rowBase  = blockIdx.x * ROWS_PER_BLOCK + wave * 16;
    const int diagTile = rowBase >> 4;
    const int ctBegin  = blockIdx.y * TILES_PER_CHUNK;
    const int ctEnd    = ctBegin + TILES_PER_CHUNK;
    const bool hasDiag = (diagTile >= ctBegin) && (diagTile < ctEnd);

    // K offsets of the 8 dword-pairs of a 16x32 f16 A/B fragment (ISA §7.12.2)
    int kofs[8];
    #pragma unroll
    for (int p = 0; p < 8; ++p)
        kofs[p] = (p < 4) ? (hh * 8 + 2 * p) : (16 + hh * 8 + 2 * (p - 4));

    // A fragments: this wave's 16 rows x K=256, loaded once, kept in VGPRs.
    FragH afrag[8];
    const _Float16* arow = zn + (size_t)(rowBase + lm) * DIM;
    #pragma unroll
    for (int kb = 0; kb < 8; ++kb)
        #pragma unroll
        for (int p = 0; p < 8; ++p)
            afrag[kb].u[p] = *(const uint32_t*)(arow + kb * 32 + kofs[p]);

    // Self-dot of row (rowBase+lm) in f16 (for the diagonal correction):
    // lanes (0,lm) and (1,lm) each hold half of the 256 K values.
    float sd = 0.0f;
    #pragma unroll
    for (int kb = 0; kb < 8; ++kb)
        #pragma unroll
        for (int e = 0; e < 16; ++e) {
            float x = (float)afrag[kb].h[e];
            sd = fmaf(x, x, sd);
        }
    sd += __shfl_xor(sd, 16, 32);   // combine the two K halves

    float s[8];
    #pragma unroll
    for (int v = 0; v < 8; ++v) s[v] = 0.0f;

    const uint32_t ldsBase = (uint32_t)(uintptr_t)&ldsB[0];
    const int idx0 = tid;           // this thread's two 16B slices of the 8KB tile
    const int idx1 = tid + 256;
    const uint32_t loff0 = ldsBase + (uint32_t)((idx0 >> 5) * (LDS_STRIDE * 2) + (idx0 & 31) * 16);
    const uint32_t loff1 = ldsBase + (uint32_t)((idx1 >> 5) * (LDS_STRIDE * 2) + (idx1 & 31) * 16);

    for (int ct = ctBegin; ct < ctEnd; ++ct) {
        const char* gbase = (const char*)zn + (size_t)ct * 16 * DIM * 2;
        if (ct + 1 < ctEnd)   // warm L2 for next tile (global_prefetch_b8)
            __builtin_prefetch((const void*)(gbase + 16 * DIM * 2 + tid * 8), 0, 0);

        __syncthreads();      // previous iteration's LDS reads done
        {
            uint64_t g0 = (uint64_t)(uintptr_t)(gbase + idx0 * 16);
            uint64_t g1 = (uint64_t)(uintptr_t)(gbase + idx1 * 16);
            asm volatile("global_load_async_to_lds_b128 %0, %1, off"
                         :: "v"(loff0), "v"(g0) : "memory");
            asm volatile("global_load_async_to_lds_b128 %0, %1, off"
                         :: "v"(loff1), "v"(g1) : "memory");
            asm volatile("s_wait_asynccnt 0x0" ::: "memory");
        }
        __syncthreads();

        // Load all 8 B fragments first so ds_loads overlap the WMMA chain.
        const _Float16* brow = ldsB + lm * LDS_STRIDE;  // column (ct*16+lm)
        FragH bf[8];
        #pragma unroll
        for (int kb = 0; kb < 8; ++kb)
            #pragma unroll
            for (int p = 0; p < 8; ++p)
                bf[kb].u[p] = *(const uint32_t*)(brow + kb * 32 + kofs[p]);

        v8f c8 = {0.f, 0.f, 0.f, 0.f, 0.f, 0.f, 0.f, 0.f};
        #pragma unroll
        for (int kb = 0; kb < 8; ++kb)
            c8 = __builtin_amdgcn_wmma_f32_16x16x32_f16(
                     false, afrag[kb].h, false, bf[kb].h, (short)0, c8, false, false);

        // C layout: VGPR v, half hh -> row (v + 8*hh); lane%16 -> column.
        // Branch-free: one FMA + one v_exp per element (diag fixed in the tail).
        #pragma unroll
        for (int v = 0; v < 8; ++v)
            s[v] += __builtin_amdgcn_exp2f(fmaf(c8[v], EXP2_SCALE, EXP2_BIAS));
    }

    // Combine partial sums across the 16 lanes that share each row.
    #pragma unroll
    for (int v = 0; v < 8; ++v) {
        float t = s[v];
        for (int off = 1; off < 16; off <<= 1) t += __shfl_xor(t, off, 16);
        s[v] = t;
    }
    if (lm == 0) {
        float* dst = psum + (size_t)blockIdx.y * N_TOT;
        #pragma unroll
        for (int v = 0; v < 8; ++v) {
            // diagonal correction: subtract exp(selfdot/T - 2) in the owning chunk
            float sdv  = __shfl(sd, v + 8 * hh, 16);
            float diag = hasDiag ? __builtin_amdgcn_exp2f(fmaf(sdv, EXP2_SCALE, EXP2_BIAS))
                                 : 0.0f;
            dst[rowBase + v + 8 * hh] = s[v] - diag;
        }
    }
}

// ---------------------------------------------------------------------------
// Kernel 4: per-row lse + softplus + mean -> d_out[0]
// ---------------------------------------------------------------------------
__global__ void __launch_bounds__(256)
ntx_finish(const float* __restrict__ psum, const float* __restrict__ pos_logit,
           float* __restrict__ out) {
    __shared__ float red[256];
    float t = 0.0f;
    for (int i = threadIdx.x; i < N_TOT; i += 256) {
        float ssum = 0.0f;
        #pragma unroll
        for (int c = 0; c < COL_CHUNKS; ++c) ssum += psum[(size_t)c * N_TOT + i];
        const float lse_neg = 2.0f + __logf(ssum);
        // loss_i = logaddexp(pos, lse_neg) - pos = softplus(lse_neg - pos)
        t += log1pf(__expf(lse_neg - pos_logit[i]));
    }
    red[threadIdx.x] = t;
    __syncthreads();
    for (int sft = 128; sft > 0; sft >>= 1) {
        if (threadIdx.x < sft) red[threadIdx.x] += red[threadIdx.x + sft];
        __syncthreads();
    }
    if (threadIdx.x == 0) out[0] = red[0] / (float)N_TOT;
}

// ---------------------------------------------------------------------------
extern "C" void kernel_launch(void* const* d_in, const int* in_sizes, int n_in,
                              void* d_out, int out_size, void* d_ws, size_t ws_size,
                              hipStream_t stream) {
    (void)in_sizes; (void)n_in; (void)out_size; (void)ws_size;
    const float* z1 = (const float*)d_in[0];
    const float* z2 = (const float*)d_in[1];
    float* out = (float*)d_out;

    char* ws = (char*)d_ws;
    _Float16* zn        = (_Float16*)(ws);                               // 4 MB
    float*    pos_logit = (float*)(ws + (size_t)N_TOT * DIM * 2);        // 32 KB
    float*    psum      = (float*)(ws + (size_t)N_TOT * DIM * 2 + N_TOT * 4); // 256 KB

    ntx_normalize<<<N_TOT / 8, 256, 0, stream>>>(z1, z2, zn);
    ntx_positives<<<N_TOT / 8, 256, 0, stream>>>(zn, pos_logit);
    dim3 grid(N_TOT / ROWS_PER_BLOCK, COL_CHUNKS);
    ntx_sim_lse<<<grid, 256, 0, stream>>>(zn, psum);
    ntx_finish<<<1, 256, 0, stream>>>(psum, pos_logit, out);
}
